// InteractionLayer_61778809585741
// MI455X (gfx1250) — matrix operand order repaired and checked by reading, not compiled, not run on hardware
//
#include <hip/hip_runtime.h>
#include <hip/hip_bf16.h>

// ---------------------------------------------------------------------------
// Problem constants (from the reference): B=128, T=64, E=2048, H=1024, R=8
// ---------------------------------------------------------------------------
static constexpr int Bc = 128;
static constexpr int Tc = 64;
static constexpr int Ec = 2048;
static constexpr int Hc = 1024;
static constexpr int Rc = 8;

typedef __attribute__((ext_vector_type(8)))  __bf16 v8bf;
typedef __attribute__((ext_vector_type(16))) __bf16 v16bf;
typedef __attribute__((ext_vector_type(8)))  float  v8f;
typedef __attribute__((__vector_size__(16))) int    v4i;   // matches builtin param type

// gfx1250 async global->LDS copy path (guarded so both toolchains compile)
#if __has_builtin(__builtin_amdgcn_global_load_async_to_lds_b128) && \
    __has_builtin(__builtin_amdgcn_s_wait_asynccnt)
#define USE_ASYNC_LDS 1
#else
#define USE_ASYNC_LDS 0
#endif

// fp32 -> bf16, round-to-nearest-even
__device__ __forceinline__ __bf16 f2bf(float f) {
    unsigned u = __builtin_bit_cast(unsigned, f);
    unsigned r = u + 0x7fffu + ((u >> 16) & 1u);
    unsigned short h = (unsigned short)(r >> 16);
    return __builtin_bit_cast(__bf16, h);
}

__device__ __forceinline__ float sigmoidf_(float x) {
    return 1.0f / (1.0f + expf(-x));
}

// ---------------------------------------------------------------------------
// fp32 -> bf16 bulk convert (weights / encoder), once per launch
// ---------------------------------------------------------------------------
__global__ void cvt_f32_bf16(const float* __restrict__ src,
                             __bf16* __restrict__ dst, int n) {
    int i = blockIdx.x * blockDim.x + threadIdx.x;
    if (i < n) dst[i] = f2bf(src[i]);
}

__global__ void zero_f32(float* __restrict__ p, int n) {
    int i = blockIdx.x * blockDim.x + threadIdx.x;
    if (i < n) p[i] = 0.0f;
}

// ---------------------------------------------------------------------------
// bf16 WMMA GEMM:  C[M,N] = concat_K(A1[M,K1], A2[M,K2]) * W[N,K]^T + bias[N]
// W row-major [N, K1+K2].
//
// Blocking: one 256-thread block (8 waves) computes a 128(M) x 64(N) strip.
// The 64-row weight strip is staged through LDS in K-slices of 128 and shared
// by all 8 waves (8x reduction of vector-memory traffic on the dominant
// operand; weights are L2-resident across the 64 recurrent steps anyway).
// Staging uses GLOBAL_LOAD_ASYNC_TO_LDS_B128 + s_wait_asynccnt when the
// builtins exist, else a plain global_load + ds_store fallback.
//
// A-operand VGPR layout per CDNA5 ISA 7.12.2 (16-bit, 16x32):
//   lanes 0-15  : K[0..7]  , K[16..23]   (row M = lane&15)
//   lanes 16-31 : K[8..15] , K[24..31]
// B-operand (32x16, K-major per lane) per the ISA B-matrix tables:
//   lanes 0-15  : K[0..7]  , K[16..23]   (col N = lane&15)
//   lanes 16-31 : K[8..15] , K[24..31]
// ---------------------------------------------------------------------------
static constexpr int KSTEP = 128;                 // K-depth staged per barrier
static constexpr int BROWS = 64;                  // weight rows per block
static constexpr int BRS   = KSTEP + 8;           // padded LDS row stride (68 dwords)

__global__ __launch_bounds__(256)
void gemm_bf16_wmma(const __bf16* __restrict__ A1, long lda1, int K1,
                    const __bf16* __restrict__ A2, long lda2, int K2,
                    const __bf16* __restrict__ W,
                    const float* __restrict__ bias,
                    float* __restrict__ C, int M, int N) {
    __shared__ __bf16 sB[BROWS * BRS];            // 64 * 136 * 2 = 17408 B

    const int tid  = threadIdx.x;
    const int lane = tid & 31;
    const int wave = tid >> 5;

    const int tilesN = N >> 6;                    // 64-wide N strips
    const int mblk   = blockIdx.x / tilesN;       // 128-tall M blocks
    const int tn     = blockIdx.x % tilesN;
    const int tm     = (mblk << 3) + wave;        // this wave's 16-row M tile

    const int lrow  = lane & 15;
    const int lhalf = (lane & 16) ? 8 : 0;        // A-operand K sub-offset
    const int bhalf = (lane & 16) ? 8 : 0;        // B-operand K sub-offset

    const int  K   = K1 + K2;
    const long ldW = (long)K;
    const int  n0  = tn << 6;

    const __bf16* Wbase = W + (long)n0 * ldW;     // 64-row weight strip

    const int arow = (tm << 4) + lrow;
    const __bf16* a1p = A1 + (long)arow * lda1 + lhalf;
    const __bf16* a2p = (A2 != nullptr) ? (A2 + (long)arow * lda2 + lhalf) : nullptr;

    v8f acc0, acc1, acc2, acc3;
    { v8f z = {}; acc0 = z; acc1 = z; acc2 = z; acc3 = z; }

    for (int k0 = 0; k0 < K; k0 += KSTEP) {
        // ---- cooperative stage: 64 rows x 128 elems (bf16) -> LDS ---------
        // 64*16 = 1024 16-byte chunks, 256 threads -> 4 chunks per thread
        #pragma unroll
        for (int c = 0; c < 4; ++c) {
            int idx = (c << 8) + tid;
            int row = idx >> 4;
            int chk = (idx & 15) << 3;            // element offset within slice
            const __bf16* src = Wbase + (long)row * ldW + k0 + chk;
            __bf16* dst = &sB[row * BRS + chk];
#if USE_ASYNC_LDS
            __builtin_amdgcn_global_load_async_to_lds_b128(
                (__attribute__((address_space(1))) v4i*)(void*)const_cast<__bf16*>(src),
                (__attribute__((address_space(3))) v4i*)(void*)dst, 0, 0);
#else
            *(v8bf*)dst = *(const v8bf*)src;
#endif
        }
        // prefetch the next weight slice while this one is consumed
        if (k0 + KSTEP < K)
            __builtin_prefetch(Wbase + (long)(lane << 1) * ldW + k0 + KSTEP, 0, 1);
#if USE_ASYNC_LDS
        __builtin_amdgcn_s_wait_asynccnt(0);
#endif
        __syncthreads();

        // ---- compute: 4 k-chunks x 4 N-subtiles = 16 WMMA per stage -------
        const __bf16* ap = (k0 < K1) ? (a1p + k0) : (a2p + (k0 - K1));  // uniform

        #pragma unroll
        for (int kk = 0; kk < KSTEP; kk += 32) {
            v8bf alo = *(const v8bf*)(ap + kk);
            v8bf ahi = *(const v8bf*)(ap + kk + 16);
            v16bf a  = __builtin_shufflevector(alo, ahi,
                         0,1,2,3,4,5,6,7,8,9,10,11,12,13,14,15);

            {
                const __bf16* bp = &sB[(0 + lrow) * BRS + kk + bhalf];
                v8bf blo = *(const v8bf*)(bp);
                v8bf bhi = *(const v8bf*)(bp + 16);
                v16bf b  = __builtin_shufflevector(blo, bhi,
                             0,1,2,3,4,5,6,7,8,9,10,11,12,13,14,15);
                acc0 = __builtin_amdgcn_wmma_f32_16x16x32_bf16(false, a, false, b,
                                                               (short)0, acc0, false, false);
            }
            {
                const __bf16* bp = &sB[(16 + lrow) * BRS + kk + bhalf];
                v8bf blo = *(const v8bf*)(bp);
                v8bf bhi = *(const v8bf*)(bp + 16);
                v16bf b  = __builtin_shufflevector(blo, bhi,
                             0,1,2,3,4,5,6,7,8,9,10,11,12,13,14,15);
                acc1 = __builtin_amdgcn_wmma_f32_16x16x32_bf16(false, a, false, b,
                                                               (short)0, acc1, false, false);
            }
            {
                const __bf16* bp = &sB[(32 + lrow) * BRS + kk + bhalf];
                v8bf blo = *(const v8bf*)(bp);
                v8bf bhi = *(const v8bf*)(bp + 16);
                v16bf b  = __builtin_shufflevector(blo, bhi,
                             0,1,2,3,4,5,6,7,8,9,10,11,12,13,14,15);
                acc2 = __builtin_amdgcn_wmma_f32_16x16x32_bf16(false, a, false, b,
                                                               (short)0, acc2, false, false);
            }
            {
                const __bf16* bp = &sB[(48 + lrow) * BRS + kk + bhalf];
                v8bf blo = *(const v8bf*)(bp);
                v8bf bhi = *(const v8bf*)(bp + 16);
                v16bf b  = __builtin_shufflevector(blo, bhi,
                             0,1,2,3,4,5,6,7,8,9,10,11,12,13,14,15);
                acc3 = __builtin_amdgcn_wmma_f32_16x16x32_bf16(false, a, false, b,
                                                               (short)0, acc3, false, false);
            }
        }
        __syncthreads();                          // protect sB before next stage
    }

    // Epilogue: C layout (ISA 7.12.2): VGPR i -> row i (lanes 0-15) / row 8+i
    // (lanes 16-31); col = lane & 15.  Add per-column bias.
    const int mbase = (tm << 4) + ((lane & 16) ? 8 : 0);
    {
        int n = n0 + 0 + lrow;  float bj = bias[n];
        #pragma unroll
        for (int i = 0; i < 8; ++i) C[(long)(mbase + i) * N + n] = acc0[i] + bj;
    }
    {
        int n = n0 + 16 + lrow; float bj = bias[n];
        #pragma unroll
        for (int i = 0; i < 8; ++i) C[(long)(mbase + i) * N + n] = acc1[i] + bj;
    }
    {
        int n = n0 + 32 + lrow; float bj = bias[n];
        #pragma unroll
        for (int i = 0; i < 8; ++i) C[(long)(mbase + i) * N + n] = acc2[i] + bj;
    }
    {
        int n = n0 + 48 + lrow; float bj = bias[n];
        #pragma unroll
        for (int i = 0; i < 8; ++i) C[(long)(mbase + i) * N + n] = acc3[i] + bj;
    }
}

// ---------------------------------------------------------------------------
// Gather role-state rows of A into bf16 GEMM operand buffers.
// blockIdx.x = b*8 + s ; s: 0=spk, 1=adr, 2..7=others (increasing role order,
// matching the reference's stable argsort).
// ---------------------------------------------------------------------------
__global__ __launch_bounds__(256)
void gather_kernel(const float* __restrict__ A, const int* __restrict__ dig, int t,
                   __bf16* __restrict__ Hspk, __bf16* __restrict__ Hadr,
                   __bf16* __restrict__ Hoth) {
    const int b = blockIdx.x >> 3;
    const int s = blockIdx.x & 7;
    const int spk = dig[((long)b * Tc + t) * 2 + 0];
    const int adr = dig[((long)b * Tc + t) * 2 + 1];

    int role;
    __bf16* dst;
    if (s == 0)      { role = spk; dst = Hspk + (long)b * Hc; }
    else if (s == 1) { role = adr; dst = Hadr + (long)b * Hc; }
    else {
        int oth[6]; int c = 0;
        #pragma unroll
        for (int r = 0; r < Rc; ++r) if (r != spk && r != adr) oth[c++] = r;
        role = oth[s - 2];
        dst = Hoth + ((long)b * 6 + (s - 2)) * Hc;
    }
    const float* src = A + ((long)b * Rc + role) * Hc;
    for (int h = threadIdx.x; h < Hc; h += 256) dst[h] = f2bf(src[h]);
}

// ---------------------------------------------------------------------------
// Fused GRU gate nonlinearity + state update (scatter into A).
// Each (b, role-slot) handled by exactly one block: reads old A, writes new.
// ---------------------------------------------------------------------------
__global__ __launch_bounds__(256)
void combine_kernel(float* __restrict__ A, const int* __restrict__ dig, int t,
                    const float* __restrict__ GIs, const float* __restrict__ GHs,
                    const float* __restrict__ GIa, const float* __restrict__ GHa,
                    const float* __restrict__ GIo, const float* __restrict__ GHo) {
    const int b = blockIdx.x >> 3;
    const int s = blockIdx.x & 7;
    const int spk = dig[((long)b * Tc + t) * 2 + 0];
    const int adr = dig[((long)b * Tc + t) * 2 + 1];

    int role;
    const float *gi, *gh;
    if (s == 0)      { role = spk; gi = GIs + (long)b * 3 * Hc; gh = GHs + (long)b * 3 * Hc; }
    else if (s == 1) { role = adr; gi = GIa + (long)b * 3 * Hc; gh = GHa + (long)b * 3 * Hc; }
    else {
        int oth[6]; int c = 0;
        #pragma unroll
        for (int r = 0; r < Rc; ++r) if (r != spk && r != adr) oth[c++] = r;
        role = oth[s - 2];
        gi = GIo + (long)b * 3 * Hc;                       // gi shared (enc broadcast)
        gh = GHo + ((long)b * 6 + (s - 2)) * 3 * Hc;
    }

    float* hv = A + ((long)b * Rc + role) * Hc;
    for (int i = threadIdx.x; i < Hc; i += 256) {
        float ir = gi[i], iz = gi[Hc + i], ig = gi[2 * Hc + i];
        float hr = gh[i], hz = gh[Hc + i], hg = gh[2 * Hc + i];
        float r  = sigmoidf_(ir + hr);
        float z  = sigmoidf_(iz + hz);
        float n  = tanhf(ig + r * hg);
        float ho = hv[i];
        hv[i] = (1.0f - z) * n + z * ho;
    }
}

// ---------------------------------------------------------------------------
// Host-side orchestration
// ---------------------------------------------------------------------------
extern "C" void kernel_launch(void* const* d_in, const int* in_sizes, int n_in,
                              void* d_out, int out_size, void* d_ws, size_t ws_size,
                              hipStream_t stream) {
    const float* enc    = (const float*)d_in[0];
    const int*   dig    = (const int*)  d_in[1];
    const float* ws_ih  = (const float*)d_in[2];
    const float* ws_hh  = (const float*)d_in[3];
    const float* ws_bih = (const float*)d_in[4];
    const float* ws_bhh = (const float*)d_in[5];
    const float* wa_ih  = (const float*)d_in[6];
    const float* wa_hh  = (const float*)d_in[7];
    const float* wa_bih = (const float*)d_in[8];
    const float* wa_bhh = (const float*)d_in[9];
    const float* wo_ih  = (const float*)d_in[10];
    const float* wo_hh  = (const float*)d_in[11];
    const float* wo_bih = (const float*)d_in[12];
    const float* wo_bhh = (const float*)d_in[13];

    float* A = (float*)d_out;               // live state; final state = output

    // ---- carve workspace --------------------------------------------------
    char*  base = (char*)d_ws;
    size_t off  = 0;
    auto carve = [&](size_t bytes) -> char* {
        char* p = base + off;
        off = (off + bytes + 255) & ~(size_t)255;
        return p;
    };
    const long nEnc  = (long)Bc * Tc * Ec;        // 16.8M
    const long nWsih = 3L * Hc * (Ec + Hc);       // 9.44M
    const long nWhh  = 3L * Hc * Hc;              // 3.15M
    const long nWoih = 3L * Hc * Ec;              // 6.29M

    __bf16* encB  = (__bf16*)carve((size_t)nEnc  * 2);
    __bf16* wsihB = (__bf16*)carve((size_t)nWsih * 2);
    __bf16* wshhB = (__bf16*)carve((size_t)nWhh  * 2);
    __bf16* waihB = (__bf16*)carve((size_t)nWsih * 2);
    __bf16* wahhB = (__bf16*)carve((size_t)nWhh  * 2);
    __bf16* woihB = (__bf16*)carve((size_t)nWoih * 2);
    __bf16* wohhB = (__bf16*)carve((size_t)nWhh  * 2);

    __bf16* Hspk = (__bf16*)carve((size_t)Bc * Hc * 2);
    __bf16* Hadr = (__bf16*)carve((size_t)Bc * Hc * 2);
    __bf16* Hoth = (__bf16*)carve((size_t)Bc * 6 * Hc * 2);

    float* GIs = (float*)carve((size_t)Bc * 3 * Hc * 4);
    float* GHs = (float*)carve((size_t)Bc * 3 * Hc * 4);
    float* GIa = (float*)carve((size_t)Bc * 3 * Hc * 4);
    float* GHa = (float*)carve((size_t)Bc * 3 * Hc * 4);
    float* GIo = (float*)carve((size_t)Bc * 3 * Hc * 4);
    float* GHo = (float*)carve((size_t)Bc * 6 * 3 * Hc * 4);
    (void)ws_size; (void)in_sizes; (void)n_in; (void)out_size;

    // ---- once per launch: bf16 conversions + A = 0 ------------------------
    auto cvt = [&](const float* s, __bf16* d, long n) {
        cvt_f32_bf16<<<dim3((unsigned)((n + 255) / 256)), dim3(256), 0, stream>>>(s, d, (int)n);
    };
    cvt(enc,   encB,  nEnc);
    cvt(ws_ih, wsihB, nWsih);
    cvt(ws_hh, wshhB, nWhh);
    cvt(wa_ih, waihB, nWsih);
    cvt(wa_hh, wahhB, nWhh);
    cvt(wo_ih, woihB, nWoih);
    cvt(wo_hh, wohhB, nWhh);

    {
        long nA = (long)Bc * Rc * Hc;
        zero_f32<<<dim3((unsigned)((nA + 255) / 256)), dim3(256), 0, stream>>>(A, (int)nA);
    }

    // ---- GEMM launcher ----------------------------------------------------
    auto gemm = [&](const __bf16* A1, long lda1, int K1,
                    const __bf16* A2, long lda2, int K2,
                    const __bf16* W, const float* bias,
                    float* C, int M, int N) {
        int blocks = (M >> 7) * (N >> 6);   // 128x64 strip per 8-wave block
        gemm_bf16_wmma<<<dim3(blocks), dim3(256), 0, stream>>>(
            A1, lda1, K1, A2, lda2, K2, W, bias, C, M, N);
    };

    const long ldEnc = (long)Tc * Ec;        // row stride of enc[b, t, :] over b
    const int  N3H   = 3 * Hc;               // 3072

    // ---- sequential scan over T -------------------------------------------
    for (int t = 0; t < Tc; ++t) {
        const __bf16* encT = encB + (long)t * Ec;

        gather_kernel<<<dim3(Bc * 8), dim3(256), 0, stream>>>(A, dig, t, Hspk, Hadr, Hoth);

        // spk: gi = [enc_t | adr_v] @ ws_ih^T + b ; gh = spk_v @ ws_hh^T + b
        gemm(encT, ldEnc, Ec,  Hadr, Hc, Hc,  wsihB, ws_bih, GIs, Bc, N3H);
        gemm(Hspk, Hc,  Hc,  nullptr, 0, 0,   wshhB, ws_bhh, GHs, Bc, N3H);
        // adr: gi = [enc_t | spk_v] @ wa_ih^T + b ; gh = adr_v @ wa_hh^T + b
        gemm(encT, ldEnc, Ec,  Hspk, Hc, Hc,  waihB, wa_bih, GIa, Bc, N3H);
        gemm(Hadr, Hc,  Hc,  nullptr, 0, 0,   wahhB, wa_bhh, GHa, Bc, N3H);
        // others: gi depends only on enc_t (broadcast) -> M=B once;
        //         gh over all 6 other roles -> M = B*6
        gemm(encT, ldEnc, Ec,  nullptr, 0, 0,  woihB, wo_bih, GIo, Bc, N3H);
        gemm(Hoth, Hc,  Hc,  nullptr, 0, 0,    wohhB, wo_bhh, GHo, Bc * 6, N3H);

        combine_kernel<<<dim3(Bc * 8), dim3(256), 0, stream>>>(
            A, dig, t, GIs, GHs, GIa, GHa, GIo, GHo);
    }
}
